// ScaleShiftMACE_51573967290534
// MI455X (gfx1250) — matrix operand order repaired and checked by reading, not compiled, not run on hardware
//
#include <hip/hip_runtime.h>
#include <hip/hip_bf16.h>

typedef _Float16 half_t;
typedef __attribute__((ext_vector_type(16))) _Float16 v16h;
typedef __attribute__((ext_vector_type(8)))  _Float16 v8h;
typedef __attribute__((ext_vector_type(8)))  float    v8f;

#define N_NODES 10000
#define N_EDGES 80000
#define NG 32
#define NZ 10
#define NC 128
#define NB 8
#define NH 64

// padded K-strides for transposed LDS weights (bank-conflict-free b128 loads)
#define W1T_LD 40   // K=32 padded
#define W64_LD 72   // K=64 padded
#define W128_LD 136 // K=128 padded
#define ACT_LD 72   // 16x64 activation tile, K padded

// ---------------- WMMA fragment helpers (CDNA5 ISA 7.12.2 layouts) ----------
// A 16x32 f16: lane<16 K-base 0, lane>=16 K-base 8; elements 8..15 -> K+16.
// B 32x16 f16: lane<16 K-base 0, lane>=16 K-base 8; elements 8..15 -> K+16; N = lane&15.
// C/D 16x16 f32: M = v + 8*(lane>=16); N = lane&15.
__device__ __forceinline__ int cd_m(int lane, int v) { return v + ((lane & 16) ? 8 : 0); }

__device__ __forceinline__ float silu_f(float x) {
  return x * __builtin_amdgcn_rcpf(1.0f + __expf(-x));  // v_rcp_f32, no div expansion
}

__device__ __forceinline__ void lds_fence_wave() {
  asm volatile("s_wait_dscnt 0" ::: "memory");  // CDNA5 split counter wait
}

// A fragment from row-major LDS f16 tile: two contiguous 16B loads per lane.
__device__ __forceinline__ v16h ldsA(const half_t* __restrict__ tile, int ld,
                                     int kc32, int lane) {
  const half_t* p = tile + (lane & 15) * ld + kc32 + ((lane & 16) ? 8 : 0);
  v8h lo = *(const v8h*)p;         // k .. k+7   (t=0..7)
  v8h hi = *(const v8h*)(p + 16);  // k+16..k+23 (t=8..15)
  return __builtin_shufflevector(lo, hi, 0, 1, 2, 3, 4, 5, 6, 7,
                                 8, 9, 10, 11, 12, 13, 14, 15);
}

// B fragment from transposed ([n][k]) LDS weights: two contiguous 16B loads.
template <int LDK>
__device__ __forceinline__ v16h ldsB(const half_t* __restrict__ Wt, int kbase,
                                     int nbase, int lane) {
  const half_t* p = Wt + (nbase + (lane & 15)) * LDK + kbase + ((lane & 16) ? 8 : 0);
  v8h lo = *(const v8h*)p;
  v8h hi = *(const v8h*)(p + 16);
  return __builtin_shufflevector(lo, hi, 0, 1, 2, 3, 4, 5, 6, 7,
                                 8, 9, 10, 11, 12, 13, 14, 15);
}

// A fragment from a contiguous f32 row in global memory (float4 loads + cvt).
__device__ __forceinline__ v16h f32A(const float* __restrict__ rowbase, int kc32,
                                     int lane) {
  const float* p = rowbase + kc32 + ((lane & 16) ? 8 : 0);
  float4 q0 = *(const float4*)p;
  float4 q1 = *(const float4*)(p + 4);
  float4 q2 = *(const float4*)(p + 16);
  float4 q3 = *(const float4*)(p + 20);
  v16h a;
  a[0] = (half_t)q0.x; a[1] = (half_t)q0.y; a[2] = (half_t)q0.z; a[3] = (half_t)q0.w;
  a[4] = (half_t)q1.x; a[5] = (half_t)q1.y; a[6] = (half_t)q1.z; a[7] = (half_t)q1.w;
  a[8] = (half_t)q2.x; a[9] = (half_t)q2.y; a[10] = (half_t)q2.z; a[11] = (half_t)q2.w;
  a[12] = (half_t)q3.x; a[13] = (half_t)q3.y; a[14] = (half_t)q3.z; a[15] = (half_t)q3.w;
  return a;
}

// 16x64 @ 64x64 (transposed LDS weights) -> silu -> 16x64 out tile
__device__ __forceinline__ void mlp64(const half_t* __restrict__ in,
                                      const half_t* __restrict__ Wt,
                                      half_t* __restrict__ out, int lane) {
  v16h a[2];
#pragma unroll
  for (int kc = 0; kc < 2; ++kc) a[kc] = ldsA(in, ACT_LD, kc * 32, lane);
#pragma unroll
  for (int cb = 0; cb < 4; ++cb) {
    v8f acc = {};
#pragma unroll
    for (int kc = 0; kc < 2; ++kc) {
      v16h b = ldsB<W64_LD>(Wt, kc * 32, cb * 16, lane);
      acc = __builtin_amdgcn_wmma_f32_16x16x32_f16(false, a[kc], false, b,
                                                   (short)0, acc, false, false);
    }
#pragma unroll
    for (int v = 0; v < 8; ++v)
      out[cd_m(lane, v) * ACT_LD + cb * 16 + (lane & 15)] = (half_t)silu_f(acc[v]);
  }
}

// ---------------- K0: per-edge geometry: Y[16], radial feats[8] --------------
__global__ void edge_geom(const float* __restrict__ vec, float* __restrict__ Y,
                          float* __restrict__ feats) {
  int e = blockIdx.x * blockDim.x + threadIdx.x;
  if (e >= N_EDGES) return;
  float x0 = vec[e * 3 + 0], y0 = vec[e * 3 + 1], z0 = vec[e * 3 + 2];
  float r = sqrtf(x0 * x0 + y0 * y0 + z0 * z0);
  float ir = 1.0f / r;
  float x = x0 * ir, y = y0 * ir, z = z0 * ir;
  const float s3 = 1.7320508075688772f, s5 = 2.23606797749979f, s15 = 3.872983346207417f;
  const float s70_4 = 2.0916500663351889f, s105 = 10.246950765959598f;
  const float s42_4 = 1.6201851746019651f, s7_2 = 1.3228756555322954f;
  const float s105_2 = 5.123475382979799f;
  float* Ye = Y + (size_t)e * 16;
  Ye[0] = 1.0f; Ye[1] = s3 * x; Ye[2] = s3 * y; Ye[3] = s3 * z;
  Ye[4] = s15 * x * y; Ye[5] = s15 * y * z;
  Ye[6] = 0.5f * s5 * (3.0f * z * z - 1.0f);
  Ye[7] = s15 * x * z; Ye[8] = 0.5f * s15 * (x * x - y * y);
  Ye[9] = s70_4 * (3.0f * x * x * y - y * y * y);
  Ye[10] = s105 * x * y * z;
  Ye[11] = s42_4 * y * (5.0f * z * z - 1.0f);
  Ye[12] = s7_2 * (5.0f * z * z * z - 3.0f * z);
  Ye[13] = s42_4 * x * (5.0f * z * z - 1.0f);
  Ye[14] = s105_2 * z * (x * x - y * y);
  Ye[15] = s70_4 * (x * x * x - 3.0f * x * y * y);
  float xr = r * 0.2f;
  float x5 = xr * xr; x5 = x5 * x5 * xr;  // xr^5
  float cut = (xr < 1.0f) ? (1.0f - 21.0f * x5 + 35.0f * x5 * xr - 15.0f * x5 * xr * xr) : 0.0f;
  const float pref = 0.6324555320336759f;  // sqrt(2/5)
  float* fe = feats + (size_t)e * NB;
#pragma unroll
  for (int n = 1; n <= NB; ++n)
    fe[n - 1] = pref * __sinf((float)n * 3.14159265358979323846f * r * 0.2f) * ir * cut;
}

// ---------------- K1: node init: h, e0 per node, w = attrs@U (both layers) ---
__global__ void init_nodes(const float* __restrict__ attrs, const float* __restrict__ Wemb,
                           const float* __restrict__ aen, const float* __restrict__ U,
                           float* __restrict__ h, float* __restrict__ e0n,
                           float* __restrict__ wvec, float* __restrict__ ninter) {
  __shared__ float a[NZ];
  int n = blockIdx.x;
  if (threadIdx.x < NZ) a[threadIdx.x] = attrs[n * NZ + threadIdx.x];
  __syncthreads();
  int c = threadIdx.x;
  float s = 0.0f;
#pragma unroll
  for (int z = 0; z < NZ; ++z) s += a[z] * Wemb[z * NC + c];
  h[(size_t)n * NC + c] = s;
  if (c == 0) {
    float e = 0.0f;
#pragma unroll
    for (int z = 0; z < NZ; ++z) e += a[z] * aen[z];
    e0n[n] = e;
    ninter[n] = 0.0f;
    for (int i = 0; i < 2; ++i)
      for (int l = 0; l < 4; ++l) {
        float ww = 0.0f;
#pragma unroll
        for (int z = 0; z < NZ; ++z) ww += a[z] * U[(i * NZ + z) * 4 + l];
        wvec[((size_t)i * N_NODES + n) * 4 + l] = ww;
      }
  }
}

__global__ void zero_f32(float* __restrict__ p, size_t n) {
  size_t i = (size_t)blockIdx.x * blockDim.x + threadIdx.x;
  size_t st = (size_t)gridDim.x * blockDim.x;
  for (; i < n; i += st) p[i] = 0.0f;
}

// ---------------- K3: fused edge MLP (WMMA chain) + message scatter ----------
// agg layout: [n][m16][c] -> atomic adds are lane-contiguous in c.
__global__ void __launch_bounds__(256) edge_fused(
    const float* __restrict__ Y, const float* __restrict__ feats,
    const float* __restrict__ h, const long long* __restrict__ snd,
    const long long* __restrict__ rcv, const float* __restrict__ rw1,
    const float* __restrict__ rw2, const float* __restrict__ rw3,
    const float* __restrict__ rw4, float* __restrict__ agg) {
  extern __shared__ char smem[];
  half_t* w1t = (half_t*)smem;             // [64][W1T_LD]
  half_t* w2t = w1t + 64 * W1T_LD;         // [64][W64_LD]
  half_t* w3t = w2t + 64 * W64_LD;         // [64][W64_LD]
  half_t* w4t = w3t + 64 * W64_LD;         // [512][W64_LD]
  char* pw = (char*)(w4t + 512 * W64_LD);
  const int wave = threadIdx.x >> 5;
  const int lane = threadIdx.x & 31;
  char* my = pw + wave * 5760;
  half_t* actA = (half_t*)my;              // [16][ACT_LD]
  half_t* actB = actA + 16 * ACT_LD;
  float* Yt = (float*)(actB + 16 * ACT_LD);  // [16][16]
  int* sidx = (int*)(Yt + 256);
  int* ridx = sidx + 16;

  // transposed weight staging ([n][k], zero-padded K)
  for (int i = threadIdx.x; i < 64 * W1T_LD; i += blockDim.x) {
    int n = i / W1T_LD, k = i % W1T_LD;
    w1t[i] = (k < NB) ? (half_t)rw1[k * NH + n] : (half_t)0.0f;
  }
  for (int i = threadIdx.x; i < 64 * W64_LD; i += blockDim.x) {
    int n = i / W64_LD, k = i % W64_LD;
    half_t v2 = (k < NH) ? (half_t)rw2[k * NH + n] : (half_t)0.0f;
    half_t v3 = (k < NH) ? (half_t)rw3[k * NH + n] : (half_t)0.0f;
    w2t[i] = v2; w3t[i] = v3;
  }
  for (int i = threadIdx.x; i < 512 * W64_LD; i += blockDim.x) {
    int n = i / W64_LD, k = i % W64_LD;
    w4t[i] = (k < NH) ? (half_t)rw4[k * 512 + n] : (half_t)0.0f;
  }
  __syncthreads();

  const int numTiles = (N_EDGES + 15) / 16;
  const int wstep = gridDim.x * (blockDim.x >> 5);
  for (int tile = blockIdx.x * (blockDim.x >> 5) + wave; tile < numTiles; tile += wstep) {
    const int ebase = tile * 16;
    if (lane < 16) {
      int e = ebase + lane;
      if (e < N_EDGES) { sidx[lane] = (int)snd[e]; ridx[lane] = (int)rcv[e]; }
      else             { sidx[lane] = 0;           ridx[lane] = -1; }
    }
    for (int i = lane; i < 256; i += 32) {
      int r = i >> 4, c = i & 15;
      int e = ebase + r;
      Yt[i] = (e < N_EDGES) ? Y[(size_t)e * 16 + c] : 0.0f;
    }
    lds_fence_wave();

    // ---- GEMM1: feats(16x8, zero-padded K=32) @ rw1 -> silu -> actB
    v16h a0 = {};
    if (lane < 16) {
      int e = ebase + lane;
      if (e < N_EDGES) {
        const float4* f4 = (const float4*)(feats + (size_t)e * NB);
        float4 f0 = f4[0], f1 = f4[1];
        a0[0] = (half_t)f0.x; a0[1] = (half_t)f0.y; a0[2] = (half_t)f0.z; a0[3] = (half_t)f0.w;
        a0[4] = (half_t)f1.x; a0[5] = (half_t)f1.y; a0[6] = (half_t)f1.z; a0[7] = (half_t)f1.w;
      }
    }
#pragma unroll
    for (int cb = 0; cb < 4; ++cb) {
      v16h b = ldsB<W1T_LD>(w1t, 0, cb * 16, lane);
      v8f acc = {};
      acc = __builtin_amdgcn_wmma_f32_16x16x32_f16(false, a0, false, b, (short)0,
                                                   acc, false, false);
#pragma unroll
      for (int v = 0; v < 8; ++v)
        actB[cd_m(lane, v) * ACT_LD + cb * 16 + (lane & 15)] = (half_t)silu_f(acc[v]);
    }
    lds_fence_wave();
    mlp64(actB, w2t, actA, lane);  // GEMM2 + silu
    lds_fence_wave();
    mlp64(actA, w3t, actB, lane);  // GEMM3 + silu
    lds_fence_wave();

    // ---- GEMM4: actB(16x64) @ rw4(64x512); epilogue scatters messages
    v16h a4[2];
#pragma unroll
    for (int kc = 0; kc < 2; ++kc) a4[kc] = ldsA(actB, ACT_LD, kc * 32, lane);
    for (int cb = 0; cb < 32; ++cb) {
      v8f acc = {};
#pragma unroll
      for (int kc = 0; kc < 2; ++kc) {
        v16h b = ldsB<W64_LD>(w4t, kc * 32, cb * 16, lane);
        acc = __builtin_amdgcn_wmma_f32_16x16x32_f16(false, a4[kc], false, b,
                                                     (short)0, acc, false, false);
      }
      int col = cb * 16 + (lane & 15);
      int l = col >> 7, c = col & 127;
      int moff = l * l, mcnt = 2 * l + 1;
#pragma unroll
      for (int v = 0; v < 8; ++v) {
        int mrow = cd_m(lane, v);
        int rv = ridx[mrow];
        if (rv >= 0) {
          float hs = h[(size_t)sidx[mrow] * NC + c];
          float base = acc[v] * hs * (1.0f / 16.0f);  // /AVG_NEIGH folded
          float* dst = agg + (size_t)rv * 2048 + moff * 128 + c;
          for (int m = 0; m < mcnt; ++m)
            atomicAdd(dst + m * 128, base * Yt[mrow * 16 + moff + m]);
        }
      }
    }
  }
}

// ---------------- K4: per-l channel mix GEMM + squared-sum epilogue ----------
// A[(n,m),c] = agg[n][m][c] (contiguous rows -> float4 gathers);
// inv layout [n][l][128] -> lane-coalesced atomic epilogue.
__global__ void __launch_bounds__(256) wmix_sq(const float* __restrict__ agg,
                                               const float* __restrict__ Wl,
                                               float* __restrict__ inv, int l) {
  extern __shared__ char smem[];
  half_t* Bs = (half_t*)smem;  // [d=128][k=W128_LD] transposed
  for (int j = threadIdx.x; j < 128 * 128; j += blockDim.x) {
    int c = j >> 7, d = j & 127;
    Bs[d * W128_LD + c] = (half_t)Wl[j];
  }
  __syncthreads();
  const int lane = threadIdx.x & 31, wave = threadIdx.x >> 5;
  const int mcnt = 2 * l + 1, moff = l * l;
  const int rows = N_NODES * mcnt;
  const int numTiles = (rows + 15) / 16;
  const int wstep = gridDim.x * (blockDim.x >> 5);
  for (int tile = blockIdx.x * (blockDim.x >> 5) + wave; tile < numTiles; tile += wstep) {
    int row = tile * 16 + (lane & 15);
    bool ok = row < rows;
    int nn = ok ? row / mcnt : 0;
    int mm = ok ? moff + row % mcnt : 0;
    const float* rowbase = agg + (size_t)nn * 2048 + (size_t)mm * 128;
    v16h a[4];
#pragma unroll
    for (int kc = 0; kc < 4; ++kc) a[kc] = f32A(rowbase, kc * 32, lane);
#pragma unroll
    for (int cb = 0; cb < 8; ++cb) {
      v8f acc = {};
#pragma unroll
      for (int kc = 0; kc < 4; ++kc) {
        v16h b = ldsB<W128_LD>(Bs, kc * 32, cb * 16, lane);
        acc = __builtin_amdgcn_wmma_f32_16x16x32_f16(false, a[kc], false, b,
                                                     (short)0, acc, false, false);
      }
#pragma unroll
      for (int v = 0; v < 8; ++v) {
        int r2 = tile * 16 + cd_m(lane, v);
        if (r2 < rows) {
          int n2 = r2 / mcnt;
          int d = cb * 16 + (lane & 15);
          atomicAdd(inv + (size_t)n2 * 512 + l * 128 + d, acc[v] * acc[v]);
        }
      }
    }
  }
}

// ---------------- K5: h' = (inv . w) @ Wprod + h @ Wsc (dual-accum WMMA) -----
__global__ void __launch_bounds__(256) node_update(
    const float* __restrict__ inv, const float* __restrict__ wv,
    const float* __restrict__ h, const float* __restrict__ Wprod,
    const float* __restrict__ Wsc, float* __restrict__ hnew) {
  extern __shared__ char smem[];
  half_t* Bp = (half_t*)smem;              // [128][W128_LD]
  half_t* Bc = Bp + 128 * W128_LD;
  for (int j = threadIdx.x; j < 128 * 128; j += blockDim.x) {
    int c = j >> 7, d = j & 127;
    Bp[d * W128_LD + c] = (half_t)Wprod[j];
    Bc[d * W128_LD + c] = (half_t)Wsc[j];
  }
  __syncthreads();
  const int lane = threadIdx.x & 31, wave = threadIdx.x >> 5;
  const int numTiles = (N_NODES + 15) / 16;
  const int wstep = gridDim.x * (blockDim.x >> 5);
  for (int tile = blockIdx.x * (blockDim.x >> 5) + wave; tile < numTiles; tile += wstep) {
    int row = tile * 16 + (lane & 15);
    int rowc = (row < N_NODES) ? row : 0;
    float wl[4];
    {
      const float* wr = wv + (size_t)rowc * 4;
      wl[0] = wr[0]; wl[1] = wr[1]; wl[2] = wr[2]; wl[3] = wr[3];
    }
    const float* ivrow = inv + (size_t)rowc * 512;  // [l][128]
    const float* hrow = h + (size_t)rowc * NC;
    v16h aT[4], aH[4];
#pragma unroll
    for (int kc = 0; kc < 4; ++kc) {
      aH[kc] = f32A(hrow, kc * 32, lane);
      int off = kc * 32 + ((lane & 16) ? 8 : 0);
#pragma unroll
      for (int g = 0; g < 2; ++g) {
        const float* p = ivrow + off + g * 16;
        float s8[8] = {0, 0, 0, 0, 0, 0, 0, 0};
#pragma unroll
        for (int l = 0; l < 4; ++l) {
          const float* pl = p + l * 128;
          float4 q0 = *(const float4*)pl;
          float4 q1 = *(const float4*)(pl + 4);
          s8[0] += wl[l] * q0.x; s8[1] += wl[l] * q0.y;
          s8[2] += wl[l] * q0.z; s8[3] += wl[l] * q0.w;
          s8[4] += wl[l] * q1.x; s8[5] += wl[l] * q1.y;
          s8[6] += wl[l] * q1.z; s8[7] += wl[l] * q1.w;
        }
#pragma unroll
        for (int j = 0; j < 8; ++j) aT[kc][g * 8 + j] = (half_t)s8[j];
      }
    }
#pragma unroll
    for (int cb = 0; cb < 8; ++cb) {
      v8f acc = {};
#pragma unroll
      for (int kc = 0; kc < 4; ++kc) {
        v16h bp = ldsB<W128_LD>(Bp, kc * 32, cb * 16, lane);
        acc = __builtin_amdgcn_wmma_f32_16x16x32_f16(false, aT[kc], false, bp,
                                                     (short)0, acc, false, false);
        v16h bc = ldsB<W128_LD>(Bc, kc * 32, cb * 16, lane);
        acc = __builtin_amdgcn_wmma_f32_16x16x32_f16(false, aH[kc], false, bc,
                                                     (short)0, acc, false, false);
      }
#pragma unroll
      for (int v = 0; v < 8; ++v) {
        int n2 = tile * 16 + cd_m(lane, v);
        if (n2 < N_NODES)
          hnew[(size_t)n2 * NC + cb * 16 + (lane & 15)] = acc[v];
      }
    }
  }
}

// ---------------- K6: readouts (wave-per-node) ------------------------------
__global__ void readout0(const float* __restrict__ h, const float* __restrict__ Wro0,
                         float* __restrict__ ninter) {
  int lane = threadIdx.x & 31;
  int wgl = blockIdx.x * (blockDim.x >> 5) + (threadIdx.x >> 5);
  int step = gridDim.x * (blockDim.x >> 5);
  for (int n = wgl; n < N_NODES; n += step) {
    float p = 0.0f;
    for (int c = lane; c < NC; c += 32) p += h[(size_t)n * NC + c] * Wro0[c];
    for (int off = 16; off; off >>= 1) p += __shfl_down(p, off, 32);
    if (lane == 0) ninter[n] += p;
  }
}

__global__ void readout1(const float* __restrict__ h, const float* __restrict__ W1a,
                         const float* __restrict__ W1b, float* __restrict__ ninter) {
  int lane = threadIdx.x & 31;
  int wgl = blockIdx.x * (blockDim.x >> 5) + (threadIdx.x >> 5);
  int step = gridDim.x * (blockDim.x >> 5);
  for (int n = wgl; n < N_NODES; n += step) {
    float t[16];
#pragma unroll
    for (int j = 0; j < 16; ++j) t[j] = 0.0f;
    for (int c = lane; c < NC; c += 32) {
      float hv = h[(size_t)n * NC + c];
#pragma unroll
      for (int j = 0; j < 16; ++j) t[j] += hv * W1a[c * 16 + j];
    }
#pragma unroll
    for (int j = 0; j < 16; ++j)
      for (int off = 16; off; off >>= 1) t[j] += __shfl_down(t[j], off, 32);
    if (lane == 0) {
      float s = 0.0f;
#pragma unroll
      for (int j = 0; j < 16; ++j) s += silu_f(t[j]) * W1b[j];
      ninter[n] += s;
    }
  }
}

// ---------------- K7: per-graph segment sum ---------------------------------
__global__ void finalize(const float* __restrict__ e0n, const float* __restrict__ ninter,
                         const float* __restrict__ scale, const float* __restrict__ shift,
                         const long long* __restrict__ batch, float* __restrict__ out) {
  int n = blockIdx.x * blockDim.x + threadIdx.x;
  if (n >= N_NODES) return;
  float v = e0n[n] + scale[0] * ninter[n] + shift[0];
  atomicAdd(&out[(int)batch[n]], v);
}

// ---------------- host-side orchestration -----------------------------------
extern "C" void kernel_launch(void* const* d_in, const int* in_sizes, int n_in,
                              void* d_out, int out_size, void* d_ws, size_t ws_size,
                              hipStream_t stream) {
  (void)in_sizes; (void)n_in; (void)out_size; (void)ws_size;
  const float* vectors = (const float*)d_in[0];
  const float* attrs   = (const float*)d_in[1];
  const float* Wemb    = (const float*)d_in[2];
  const float* aen     = (const float*)d_in[3];
  const float* rw1     = (const float*)d_in[4];
  const float* rw2     = (const float*)d_in[5];
  const float* rw3     = (const float*)d_in[6];
  const float* rw4     = (const float*)d_in[7];
  const float* Wmix    = (const float*)d_in[8];
  const float* U       = (const float*)d_in[9];
  const float* Wprod   = (const float*)d_in[10];
  const float* Wsc     = (const float*)d_in[11];
  const float* Wro0    = (const float*)d_in[12];
  const float* Wro1a   = (const float*)d_in[13];
  const float* Wro1b   = (const float*)d_in[14];
  const float* scale   = (const float*)d_in[15];
  const float* shift   = (const float*)d_in[16];
  const long long* eidx  = (const long long*)d_in[17];
  const long long* batch = (const long long*)d_in[18];
  const long long* snd = eidx;
  const long long* rcv = eidx + N_EDGES;
  float* out = (float*)d_out;

  char* ws = (char*)d_ws;
  auto alloc = [&](size_t bytes) -> char* {
    char* p = ws;
    ws += (bytes + 255) & ~(size_t)255;
    return p;
  };
  float* Y      = (float*)alloc((size_t)N_EDGES * 16 * 4);
  float* feats  = (float*)alloc((size_t)N_EDGES * NB * 4);
  float* hA     = (float*)alloc((size_t)N_NODES * NC * 4);
  float* hB     = (float*)alloc((size_t)N_NODES * NC * 4);
  float* agg    = (float*)alloc((size_t)N_NODES * 2048 * 4);
  float* inv    = (float*)alloc((size_t)N_NODES * 512 * 4);
  float* e0n    = (float*)alloc((size_t)N_NODES * 4);
  float* wvec   = (float*)alloc((size_t)2 * N_NODES * 4 * 4);
  float* ninter = (float*)alloc((size_t)N_NODES * 4);

  edge_geom<<<(N_EDGES + 255) / 256, 256, 0, stream>>>(vectors, Y, feats);
  init_nodes<<<N_NODES, NC, 0, stream>>>(attrs, Wemb, aen, U, hA, e0n, wvec, ninter);

  const size_t edge_lds =
      (size_t)(64 * W1T_LD + 64 * W64_LD + 64 * W64_LD + 512 * W64_LD) * 2 + 8 * 5760;
  float* hcur = hA;
  float* hnxt = hB;
  for (int i = 0; i < 2; ++i) {
    zero_f32<<<2048, 256, 0, stream>>>(agg, (size_t)N_NODES * 2048);
    zero_f32<<<512, 256, 0, stream>>>(inv, (size_t)N_NODES * 512);
    edge_fused<<<640, 256, edge_lds, stream>>>(
        Y, feats, hcur, snd, rcv,
        rw1 + (size_t)i * NB * NH, rw2 + (size_t)i * NH * NH,
        rw3 + (size_t)i * NH * NH, rw4 + (size_t)i * NH * 4 * NC, agg);
    for (int l = 0; l < 4; ++l) {
      int rows = N_NODES * (2 * l + 1);
      int tiles = (rows + 15) / 16;
      int blocks = (tiles + 7) / 8;
      wmix_sq<<<blocks, 256, 128 * W128_LD * 2, stream>>>(
          agg, Wmix + ((size_t)i * 4 + l) * NC * NC, inv, l);
    }
    {
      int tiles = (N_NODES + 15) / 16;
      int blocks = (tiles + 7) / 8;
      node_update<<<blocks, 256, 2 * 128 * W128_LD * 2, stream>>>(
          inv, wvec + (size_t)i * N_NODES * 4, hcur,
          Wprod + (size_t)i * NC * NC, Wsc + (size_t)i * NC * NC, hnxt);
    }
    if (i == 0)
      readout0<<<(N_NODES + 7) / 8, 256, 0, stream>>>(hnxt, Wro0, ninter);
    else
      readout1<<<(N_NODES + 7) / 8, 256, 0, stream>>>(hnxt, Wro1a, Wro1b, ninter);
    float* tmp = hcur; hcur = hnxt; hnxt = tmp;
  }
  zero_f32<<<1, 64, 0, stream>>>(out, (size_t)NG);
  finalize<<<(N_NODES + 255) / 256, 256, 0, stream>>>(e0n, ninter, scale, shift, batch, out);
}
